// NASLayer_11166914969654
// MI455X (gfx1250) — compile-verified, matrix operand order checked by reading, not run on hardware
//
#include <hip/hip_runtime.h>
#include <hip/hip_bf16.h>
#include <float.h>

#define N_NODES   50000
#define N_EDGES   800000
#define DIM_IN    256
#define DIM_OUT   64
#define HEADS     4
#define GEMM_COLS 512      // [ ft: 4 heads x 64 | res: 4 heads x 64 ]
#define NEG_SLOPE 0.01f

typedef __attribute__((ext_vector_type(16))) _Float16 v16h;
typedef __attribute__((ext_vector_type(8)))  _Float16 v8h;
typedef __attribute__((ext_vector_type(8)))  float    v8f;

// ---------------------------------------------------------------- helpers

__device__ __forceinline__ v16h load_frag(const _Float16* __restrict__ p) {
    // Fragment layout (16-bit A/B, wave32): this lane's 16 f16 values are
    // K = {base..base+7} (VGPR0-3) and K = {base+16..base+23} (VGPR4-7),
    // where base already includes the lane-half offset (+8 for lanes 16-31).
    v8h lo = *(const v8h*)(p);
    v8h hi = *(const v8h*)(p + 16);
    v16h r;
#pragma unroll
    for (int i = 0; i < 8; ++i) { r[i] = lo[i]; r[i + 8] = hi[i]; }
    return r;
}

__device__ __forceinline__ void atomicMaxF(float* addr, float v) {
    // Works with init = -FLT_MAX: positive floats compare correctly as signed
    // ints; negative floats compare inverted as unsigned ints.
    if (v >= 0.0f) atomicMax((int*)addr, __float_as_int(v));
    else           atomicMin((unsigned int*)addr, __float_as_uint(v));
}

__device__ __forceinline__ float leaky(float z) {
    return z > 0.0f ? z : NEG_SLOPE * z;
}

// ---------------------------------------------------------------- casts

__global__ void k_cast_feat(const float* __restrict__ f, _Float16* __restrict__ f16) {
    int t = blockIdx.x * 256 + threadIdx.x;
    if (t < N_NODES * DIM_IN) f16[t] = (_Float16)f[t];
}

// WT16[col][k] = W(k, col), col<256 -> W_fc[h][k][o], col>=256 -> W_res[h][k][o]
__global__ void k_pack_w(const float* __restrict__ Wfc, const float* __restrict__ Wres,
                         _Float16* __restrict__ WT16) {
    int t = blockIdx.x * 256 + threadIdx.x;       // t = col*256 + k
    if (t >= GEMM_COLS * DIM_IN) return;
    int col = t >> 8;
    int k   = t & 255;
    float v;
    if (col < HEADS * DIM_OUT) {
        int h = col >> 6, o = col & 63;
        v = Wfc[h * DIM_IN * DIM_OUT + k * DIM_OUT + o];
    } else {
        int c2 = col - HEADS * DIM_OUT;
        int h = c2 >> 6, o = c2 & 63;
        v = Wres[h * DIM_IN * DIM_OUT + k * DIM_OUT + o];
    }
    WT16[t] = (_Float16)v;
}

// ---------------------------------------------------------------- WMMA GEMM
// out[N_NODES x 512] = feat16[N_NODES x 256] * W[256 x 512]
// grid.x = 50000/16 = 3125 blocks, 256 threads = 8 waves; wave w owns columns
// [w*64, w*64+64) of one 16-row tile. 8 K-steps of 32, 4 WMMAs per step.

__global__ void __launch_bounds__(256)
k_gemm_wmma(const _Float16* __restrict__ A16, const _Float16* __restrict__ BT16,
            float* __restrict__ out) {
    const int wave   = threadIdx.x >> 5;          // 0..7 -> column supertile
    const int lane   = threadIdx.x & 31;
    const int lanelo = lane & 15;
    const int kh8    = (lane >> 4) << 3;          // +8 K offset for hi half-wave
    const int mt     = blockIdx.x;
    const int ncb    = wave * 64;

    const _Float16* __restrict__ arow = A16 + (size_t)(mt * 16 + lanelo) * DIM_IN;
    const _Float16* __restrict__ b0   = BT16 + (size_t)(ncb +  0 + lanelo) * DIM_IN;
    const _Float16* __restrict__ b1   = BT16 + (size_t)(ncb + 16 + lanelo) * DIM_IN;
    const _Float16* __restrict__ b2   = BT16 + (size_t)(ncb + 32 + lanelo) * DIM_IN;
    const _Float16* __restrict__ b3   = BT16 + (size_t)(ncb + 48 + lanelo) * DIM_IN;

    v8f acc0 = {}; v8f acc1 = {}; v8f acc2 = {}; v8f acc3 = {};

#pragma unroll
    for (int kb = 0; kb < DIM_IN; kb += 32) {
        const int ko = kb + kh8;
        v16h a  = load_frag(arow + ko);
        v16h f0 = load_frag(b0 + ko);
        v16h f1 = load_frag(b1 + ko);
        v16h f2 = load_frag(b2 + ko);
        v16h f3 = load_frag(b3 + ko);
        acc0 = __builtin_amdgcn_wmma_f32_16x16x32_f16(false, a, false, f0, (short)0, acc0, false, false);
        acc1 = __builtin_amdgcn_wmma_f32_16x16x32_f16(false, a, false, f1, (short)0, acc1, false, false);
        acc2 = __builtin_amdgcn_wmma_f32_16x16x32_f16(false, a, false, f2, (short)0, acc2, false, false);
        acc3 = __builtin_amdgcn_wmma_f32_16x16x32_f16(false, a, false, f3, (short)0, acc3, false, false);
    }

    // C/D layout: lane<16 -> M = r, lane>=16 -> M = 8+r; N = ncb + t*16 + lanelo
    const int mb = mt * 16 + ((lane >> 4) << 3);
    float* __restrict__ op = out + (size_t)mb * GEMM_COLS + ncb + lanelo;
#pragma unroll
    for (int r = 0; r < 8; ++r) {
        op[(size_t)r * GEMM_COLS +  0] = acc0[r];
        op[(size_t)r * GEMM_COLS + 16] = acc1[r];
        op[(size_t)r * GEMM_COLS + 32] = acc2[r];
        op[(size_t)r * GEMM_COLS + 48] = acc3[r];
    }
}

// ---------------------------------------------------------------- attention prep

// a1[h*N+n] = sum_o ft[n][h*64+o] * Wal[h*64+o]   (ditto a2 with War)
__global__ void k_attn_logits(const float* __restrict__ gemm,
                              const float* __restrict__ Wal, const float* __restrict__ War,
                              float* __restrict__ a1, float* __restrict__ a2) {
    int t = blockIdx.x * 256 + threadIdx.x;
    if (t >= N_NODES * HEADS) return;
    int n = t >> 2, h = t & 3;
    const float* ft = gemm + (size_t)n * GEMM_COLS + h * DIM_OUT;
    const float* wl = Wal + h * DIM_OUT;
    const float* wr = War + h * DIM_OUT;
    float s1 = 0.f, s2 = 0.f;
#pragma unroll 8
    for (int o = 0; o < DIM_OUT; ++o) {
        float v = ft[o];
        s1 += v * wl[o];
        s2 += v * wr[o];
    }
    a1[h * N_NODES + n] = s1;
    a2[h * N_NODES + n] = s2;
}

__global__ void k_init(float* __restrict__ out, float* __restrict__ m,
                       float* __restrict__ denom) {
    int t = blockIdx.x * 256 + threadIdx.x;
    if (t < N_NODES * DIM_IN) out[t] = 0.f;
    if (t < HEADS * N_NODES) { m[t] = -FLT_MAX; denom[t] = 0.f; }
}

// ---------------------------------------------------------------- edge passes

__global__ void k_edge_max(const int* __restrict__ src, const int* __restrict__ dst,
                           const float* __restrict__ a1, const float* __restrict__ a2,
                           float* __restrict__ m) {
    int t = blockIdx.x * 256 + threadIdx.x;
    if (t >= N_EDGES * HEADS) return;
    int e = t >> 2, h = t & 3;
    int d = dst[e], s = src[e];
    float z = leaky(a1[h * N_NODES + d] + a2[h * N_NODES + s]);
    atomicMaxF(&m[h * N_NODES + d], z);
}

__global__ void k_edge_denom(const int* __restrict__ src, const int* __restrict__ dst,
                             const float* __restrict__ a1, const float* __restrict__ a2,
                             const float* __restrict__ m, float* __restrict__ denom) {
    int t = blockIdx.x * 256 + threadIdx.x;
    if (t >= N_EDGES * HEADS) return;
    int e = t >> 2, h = t & 3;
    int d = dst[e], s = src[e];
    float z = leaky(a1[h * N_NODES + d] + a2[h * N_NODES + s]);
    float ez = __expf(z - m[h * N_NODES + d]);
    atomicAdd(&denom[h * N_NODES + d], ez);
}

// 16 lanes per (edge, head): float4 gather of ft[src], 4 atomic f32 adds into out[dst]
__global__ void k_edge_aggregate(const int* __restrict__ src, const int* __restrict__ dst,
                                 const float* __restrict__ a1, const float* __restrict__ a2,
                                 const float* __restrict__ m, const float* __restrict__ denom,
                                 const float* __restrict__ gemm, float* __restrict__ out) {
    long long t = (long long)blockIdx.x * 256 + threadIdx.x;
    if (t >= (long long)N_EDGES * HEADS * 16) return;
    int q  = (int)(t & 15);
    int eh = (int)(t >> 4);
    int e  = eh >> 2, h = eh & 3;
    int d = dst[e], s = src[e];
    float z  = leaky(a1[h * N_NODES + d] + a2[h * N_NODES + s]);
    float ez = __expf(z - m[h * N_NODES + d]);
    float w  = ez / fmaxf(denom[h * N_NODES + d], 1e-9f);
    const float4 fv = *(const float4*)(gemm + (size_t)s * GEMM_COLS + h * DIM_OUT + q * 4);
    float* op = out + (size_t)d * (HEADS * DIM_OUT) + h * DIM_OUT + q * 4;
    atomicAdd(op + 0, w * fv.x);
    atomicAdd(op + 1, w * fv.y);
    atomicAdd(op + 2, w * fv.z);
    atomicAdd(op + 3, w * fv.w);
}

// out[n][j] = relu(accum + res[n][j]),  res = gemm column 256 + j
__global__ void k_finalize(const float* __restrict__ gemm, float* __restrict__ out) {
    int t = blockIdx.x * 256 + threadIdx.x;
    if (t >= N_NODES * HEADS * DIM_OUT) return;
    int n = t >> 8, j = t & 255;
    float v = out[t] + gemm[(size_t)n * GEMM_COLS + (HEADS * DIM_OUT) + j];
    out[t] = fmaxf(v, 0.0f);
}

// ---------------------------------------------------------------- launch

extern "C" void kernel_launch(void* const* d_in, const int* in_sizes, int n_in,
                              void* d_out, int out_size, void* d_ws, size_t ws_size,
                              hipStream_t stream) {
    const float* features = (const float*)d_in[0];
    const int*   src      = (const int*)  d_in[1];
    const int*   dst      = (const int*)  d_in[2];
    const float* W_fc     = (const float*)d_in[3];
    const float* W_al     = (const float*)d_in[4];
    const float* W_ar     = (const float*)d_in[5];
    const float* W_res    = (const float*)d_in[6];
    float*       out      = (float*)d_out;

    // workspace carve-up (all 256B-aligned sizes)
    char* ws = (char*)d_ws;
    _Float16* feat16 = (_Float16*)ws;                      ws += (size_t)N_NODES * DIM_IN * 2;   // 25.6 MB
    _Float16* WT16   = (_Float16*)ws;                      ws += (size_t)GEMM_COLS * DIM_IN * 2; // 0.25 MB
    float*    gemm   = (float*)ws;                         ws += (size_t)N_NODES * GEMM_COLS * 4;// 102.4 MB
    float*    a1     = (float*)ws;                         ws += (size_t)HEADS * N_NODES * 4;
    float*    a2     = (float*)ws;                         ws += (size_t)HEADS * N_NODES * 4;
    float*    mmax   = (float*)ws;                         ws += (size_t)HEADS * N_NODES * 4;
    float*    denom  = (float*)ws;                         ws += (size_t)HEADS * N_NODES * 4;

    const int B = 256;
    // 1. casts
    k_cast_feat<<<(N_NODES * DIM_IN + B - 1) / B, B, 0, stream>>>(features, feat16);
    k_pack_w  <<<(GEMM_COLS * DIM_IN + B - 1) / B, B, 0, stream>>>(W_fc, W_res, WT16);
    // 2. fused [ft|res] GEMM via WMMA (50000 = 3125 * 16 exactly)
    k_gemm_wmma<<<N_NODES / 16, B, 0, stream>>>(feat16, WT16, gemm);
    // 3. attention logits + init
    k_attn_logits<<<(N_NODES * HEADS + B - 1) / B, B, 0, stream>>>(gemm, W_al, W_ar, a1, a2);
    k_init<<<(N_NODES * DIM_IN + B - 1) / B, B, 0, stream>>>(out, mmax, denom);
    // 4. segment softmax (max, exp-sum) + weighted scatter aggregation
    k_edge_max  <<<(N_EDGES * HEADS + B - 1) / B, B, 0, stream>>>(src, dst, a1, a2, mmax);
    k_edge_denom<<<(N_EDGES * HEADS + B - 1) / B, B, 0, stream>>>(src, dst, a1, a2, mmax, denom);
    long long aggT = (long long)N_EDGES * HEADS * 16;
    k_edge_aggregate<<<(unsigned)((aggT + B - 1) / B), B, 0, stream>>>(src, dst, a1, a2, mmax,
                                                                      denom, gemm, out);
    // 5. residual + relu
    k_finalize<<<(N_NODES * HEADS * DIM_OUT + B - 1) / B, B, 0, stream>>>(gemm, out);
}
